// Encoder_16028817948752
// MI455X (gfx1250) — matrix operand order repaired and checked by reading, not compiled, not run on hardware
//
#include <hip/hip_runtime.h>
#include <hip/hip_bf16.h>
#include <stdint.h>

// ---------------------------------------------------------------------------
// Problem constants (from reference)
// ---------------------------------------------------------------------------
#define N_BUILDING 30
#define D_MODEL    256
#define B_GRAPHS   2048
#define EDGES_PG   240
#define N_NODES    (B_GRAPHS * N_BUILDING)   // 61440
#define N_EDGES    (B_GRAPHS * EDGES_PG)     // 491520

// WMMA vector types (probe-confirmed signatures)
typedef __bf16 v16bf __attribute__((ext_vector_type(16)));
typedef float  v8f   __attribute__((ext_vector_type(8)));

union VFrag { v16bf v; unsigned int u[8]; };

static __device__ __forceinline__ unsigned short f2bf(float f) {
    unsigned int x = __float_as_uint(f);
    unsigned int r = (x + 0x7fffu + ((x >> 16) & 1u)) >> 16;   // RNE
    return (unsigned short)r;
}

// ---------------------------------------------------------------------------
// CDNA5 async global->LDS copy (ASYNCcnt path), 16 bytes per lane.
// ---------------------------------------------------------------------------
static __device__ __forceinline__ void async_stage16(const unsigned short* gsrc,
                                                     unsigned short* ldst) {
    unsigned lds = (unsigned)(size_t)(void*)ldst;            // low 32 bits = LDS offset
    unsigned long long ga = (unsigned long long)(size_t)gsrc;
    asm volatile("global_load_async_to_lds_b128 %0, %1, off"
                 :: "v"(lds), "v"(ga) : "memory");
}
static __device__ __forceinline__ void wait_async0() {
    asm volatile("s_wait_asynccnt 0x0" ::: "memory");
}

// ---------------------------------------------------------------------------
// Small helper kernels
// ---------------------------------------------------------------------------
__global__ void k_zero_f32(float* p, int n) {
    int i = blockIdx.x * blockDim.x + threadIdx.x;
    if (i < n) p[i] = 0.0f;
}
__global__ void k_zero_i32(int* p, int n) {
    int i = blockIdx.x * blockDim.x + threadIdx.x;
    if (i < n) p[i] = 0;
}
__global__ void k_count(const int* dst, int* cnt, int e) {
    int i = blockIdx.x * blockDim.x + threadIdx.x;
    if (i < e) atomicAdd(&cnt[dst[i]], 1);
}
// W [K,N=256] f32 (row-major)  ->  Wt [N][K] bf16 (ushort)
__global__ void k_transpose_bf16(const float* __restrict__ W,
                                 unsigned short* __restrict__ Wt, int K) {
    int i = blockIdx.x * blockDim.x + threadIdx.x;
    if (i >= K * D_MODEL) return;
    int k = i / D_MODEL, n = i % D_MODEL;
    Wt[(size_t)n * K + k] = f2bf(W[(size_t)k * D_MODEL + n]);
}
__global__ void k_f2bf_arr(const float* __restrict__ in,
                           unsigned short* __restrict__ out, int n) {
    int i = blockIdx.x * blockDim.x + threadIdx.x;
    if (i < n) out[i] = f2bf(in[i]);
}
// enc[n, 0:256]  = geometry[n] @ W_geo + b_geo ; enc[n, 256:512] = emb[semantic[n]]
__global__ void k_prep_enc(const float* __restrict__ geom,
                           const int* __restrict__ sem,
                           const float* __restrict__ Wgeo,
                           const float* __restrict__ bgeo,
                           const float* __restrict__ emb,
                           unsigned short* __restrict__ encb) {
    int i = blockIdx.x * blockDim.x + threadIdx.x;
    if (i >= N_NODES * 512) return;
    int n = i >> 9, c = i & 511;
    float v;
    if (c < 256) {
        const float* g = geom + (size_t)n * 5;
        v = bgeo[c];
        #pragma unroll
        for (int j = 0; j < 5; ++j) v += g[j] * Wgeo[j * 256 + c];
    } else {
        v = emb[(size_t)sem[n] * 256 + (c - 256)];
    }
    encb[i] = f2bf(v);
}
// h = relu(cnt ? s/cnt + b : 0);  write f32 + bf16 mirrors
__global__ void k_msg_fin(const float* __restrict__ s, const int* __restrict__ cnt,
                          const float* __restrict__ b,
                          float* __restrict__ h, unsigned short* __restrict__ hb) {
    int i = blockIdx.x * blockDim.x + threadIdx.x;
    if (i >= N_NODES * D_MODEL) return;
    int n = i >> 8, c = i & 255;
    float cf = (float)cnt[n];
    float v = 0.0f;
    if (cf > 0.0f) v = s[i] / cf + b[c];
    v = v > 0.0f ? v : 0.0f;
    h[i] = v;
    hb[i] = f2bf(v);
}
// segment max over the 30 nodes of each graph -> g[:, col_off:col_off+256]
__global__ void k_segmax(const float* __restrict__ h, float* __restrict__ g, int col_off) {
    int i = blockIdx.x * blockDim.x + threadIdx.x;
    if (i >= B_GRAPHS * D_MODEL) return;
    int gr = i >> 8, c = i & 255;
    const float* p = h + (size_t)gr * N_BUILDING * D_MODEL + c;
    float m = p[0];
    #pragma unroll
    for (int j = 1; j < N_BUILDING; ++j) m = fmaxf(m, p[j * D_MODEL]);
    g[(size_t)gr * 1024 + col_off + c] = m;
}

// ---------------------------------------------------------------------------
// WMMA fragment builders (contiguous 2x16B runs -> b128 loads)
// ---------------------------------------------------------------------------
static __device__ __forceinline__ v16bf load_a_frag(const unsigned short* rowBase,
                                                    int half) {
    VFrag a;
    #pragma unroll
    for (int v = 0; v < 8; ++v) {
        int kk = half * 16 + 2 * v;
        a.u[v] = *(const unsigned int*)(rowBase + kk);
    }
    return a.v;
}
static __device__ __forceinline__ v16bf load_b_frag(const unsigned short* wp) {
    VFrag b;
    #pragma unroll
    for (int j = 0; j < 8; ++j)
        b.u[j] = *(const unsigned int*)(wp + 2 * j);
    return b.v;
}

// ---------------------------------------------------------------------------
// Dense GEMM: out[M,256] = A[M,K](bf16) @ Wt(bf16,[256][K]) (+bias)(+posBias)(+relu)
// Block: 64 rows x 256 cols, 8 waves (32-col strip each), K-loop by 32.
// Fully unrolled (KCHUNKS compile-time): async-LDS double-buffered A staging,
// B fragments pipelined one chunk ahead; SSA rotation => no v_mov/v_nop stalls.
// ---------------------------------------------------------------------------
template <int KCHUNKS>
__global__ __launch_bounds__(256) void k_gemm_dense(
    const unsigned short* __restrict__ A,
    const unsigned short* __restrict__ Wt,
    const float* __restrict__ bias, const float* __restrict__ posBias, int relu,
    float* __restrict__ outf, unsigned short* __restrict__ outb)
{
    constexpr int K = KCHUNKS * 32;
    __shared__ __align__(16) unsigned short As[2][64][32];
    const int lane = threadIdx.x & 31;
    const int wave = threadIdx.x >> 5;
    const int half = lane >> 4;
    const int lm   = lane & 15;
    const int abase = blockIdx.x * 64;
    const int nbase = wave * 32;

    v8f zero = {0.f, 0.f, 0.f, 0.f, 0.f, 0.f, 0.f, 0.f};
    v8f acc[4][2];
    #pragma unroll
    for (int mt = 0; mt < 4; ++mt) { acc[mt][0] = zero; acc[mt][1] = zero; }

    const int srow = threadIdx.x >> 2;   // 0..63
    const int spart = threadIdx.x & 3;   // 0..3 (16B each)

    const unsigned short* aRow = A + (size_t)(abase + srow) * K + spart * 8;
    const unsigned short* wp0 = Wt + (size_t)(nbase + lm)      * K + half * 16;
    const unsigned short* wp1 = Wt + (size_t)(nbase + 16 + lm) * K + half * 16;

    // prologue: stage chunk 0, preload B fragments for chunk 0
    async_stage16(aRow, &As[0][srow][spart * 8]);
    v16bf bcur0 = load_b_frag(wp0);
    v16bf bcur1 = load_b_frag(wp1);

    #pragma unroll
    for (int kc = 0; kc < KCHUNKS; ++kc) {
        const int buf = kc & 1;
        wait_async0();          // this wave's stage of chunk kc landed in LDS
        __syncthreads();        // all waves staged kc; all reads of buf^1 done

        if (kc + 1 < KCHUNKS)
            async_stage16(aRow + (kc + 1) * 32, &As[buf ^ 1][srow][spart * 8]);

        v16bf bn0 = bcur0, bn1 = bcur1;
        if (kc + 1 < KCHUNKS) {
            bn0 = load_b_frag(wp0 + (kc + 1) * 32);
            bn1 = load_b_frag(wp1 + (kc + 1) * 32);
        }

        v16bf afr[4];
        #pragma unroll
        for (int mt = 0; mt < 4; ++mt)
            afr[mt] = load_a_frag(&As[buf][mt * 16 + lm][0], half);

        #pragma unroll
        for (int mt = 0; mt < 4; ++mt) {
            acc[mt][0] = __builtin_amdgcn_wmma_f32_16x16x32_bf16(
                false, afr[mt], false, bcur0, (short)0, acc[mt][0], false, false);
            acc[mt][1] = __builtin_amdgcn_wmma_f32_16x16x32_bf16(
                false, afr[mt], false, bcur1, (short)0, acc[mt][1], false, false);
        }
        bcur0 = bn0; bcur1 = bn1;
    }

    // epilogue: bias + positional row of W_lot + relu; f32 and bf16 outputs
    #pragma unroll
    for (int mt = 0; mt < 4; ++mt) {
        #pragma unroll
        for (int nt = 0; nt < 2; ++nt) {
            const int n = nbase + nt * 16 + lm;
            #pragma unroll
            for (int r = 0; r < 8; ++r) {
                const int row = abase + mt * 16 + half * 8 + r;
                float v = acc[mt][nt][r];
                if (bias)    v += bias[n];
                if (posBias) v += posBias[(row % N_BUILDING) * 256 + n];
                if (relu)    v = v > 0.0f ? v : 0.0f;
                if (outf) outf[(size_t)row * 256 + n] = v;
                if (outb) outb[(size_t)row * 256 + n] = f2bf(v);
            }
        }
    }
}

// ---------------------------------------------------------------------------
// Message-passing GEMM: msgs[e,:] = [h[dst[e]] | h[src[e]]] @ Wt (K = 512),
// atomically accumulated into s[dst[e], :]. Fully unrolled 16-chunk pipeline;
// the A staging gathers rows through dst (k<256) / src (k>=256).
// ---------------------------------------------------------------------------
__global__ __launch_bounds__(256) void k_gemm_msg(
    const unsigned short* __restrict__ hb,
    const unsigned short* __restrict__ Wt,
    const int* __restrict__ dstIdx, const int* __restrict__ srcIdx,
    float* __restrict__ s)
{
    constexpr int K = 512;
    __shared__ __align__(16) unsigned short As[2][64][32];
    const int lane = threadIdx.x & 31;
    const int wave = threadIdx.x >> 5;
    const int half = lane >> 4;
    const int lm   = lane & 15;
    const int ebase = blockIdx.x * 64;
    const int nbase = wave * 32;

    v8f zero = {0.f, 0.f, 0.f, 0.f, 0.f, 0.f, 0.f, 0.f};
    v8f acc[4][2];
    #pragma unroll
    for (int mt = 0; mt < 4; ++mt) { acc[mt][0] = zero; acc[mt][1] = zero; }

    const int srow = threadIdx.x >> 2;
    const int spart = threadIdx.x & 3;
    const int e = ebase + srow;
    const int dnode = dstIdx[e];
    const int snode = srcIdx[e];
    const unsigned short* dRow = hb + (size_t)dnode * 256 + spart * 8;
    const unsigned short* sRow = hb + (size_t)snode * 256 + spart * 8;

    const unsigned short* wp0 = Wt + (size_t)(nbase + lm)      * K + half * 16;
    const unsigned short* wp1 = Wt + (size_t)(nbase + 16 + lm) * K + half * 16;

    // prologue: stage chunk 0 (from h[dst]), preload B fragments for chunk 0
    async_stage16(dRow, &As[0][srow][spart * 8]);
    v16bf bcur0 = load_b_frag(wp0);
    v16bf bcur1 = load_b_frag(wp1);

    #pragma unroll
    for (int kc = 0; kc < 16; ++kc) {
        const int buf = kc & 1;
        wait_async0();
        __syncthreads();

        if (kc + 1 < 16) {
            const unsigned short* g =
                ((kc + 1) < 8 ? dRow : sRow) + ((kc + 1) & 7) * 32;
            async_stage16(g, &As[buf ^ 1][srow][spart * 8]);
        }

        v16bf bn0 = bcur0, bn1 = bcur1;
        if (kc + 1 < 16) {
            bn0 = load_b_frag(wp0 + (kc + 1) * 32);
            bn1 = load_b_frag(wp1 + (kc + 1) * 32);
        }

        v16bf afr[4];
        #pragma unroll
        for (int mt = 0; mt < 4; ++mt)
            afr[mt] = load_a_frag(&As[buf][mt * 16 + lm][0], half);

        #pragma unroll
        for (int mt = 0; mt < 4; ++mt) {
            acc[mt][0] = __builtin_amdgcn_wmma_f32_16x16x32_bf16(
                false, afr[mt], false, bcur0, (short)0, acc[mt][0], false, false);
            acc[mt][1] = __builtin_amdgcn_wmma_f32_16x16x32_bf16(
                false, afr[mt], false, bcur1, (short)0, acc[mt][1], false, false);
        }
        bcur0 = bn0; bcur1 = bn1;
    }

    // scatter epilogue: mean-aggregate numerator via f32 atomics (L2)
    #pragma unroll
    for (int mt = 0; mt < 4; ++mt) {
        #pragma unroll
        for (int r = 0; r < 8; ++r) {
            const int ee = ebase + mt * 16 + half * 8 + r;
            const int d = dstIdx[ee];
            #pragma unroll
            for (int nt = 0; nt < 2; ++nt) {
                const int n = nbase + nt * 16 + lm;
                atomicAdd(&s[(size_t)d * 256 + n], acc[mt][nt][r]);
            }
        }
    }
}

// ---------------------------------------------------------------------------
// Host-side orchestration
// ---------------------------------------------------------------------------
static inline char* wsalloc(void* ws, size_t& off, size_t bytes) {
    char* p = (char*)ws + off;
    off += (bytes + 255) & ~(size_t)255;
    return p;
}

extern "C" void kernel_launch(void* const* d_in, const int* in_sizes, int n_in,
                              void* d_out, int out_size, void* d_ws, size_t ws_size,
                              hipStream_t stream) {
    (void)in_sizes; (void)n_in; (void)out_size; (void)ws_size;

    const float* geometry = (const float*)d_in[0];
    const int*   semantic = (const int*)d_in[1];
    const int*   edge_idx = (const int*)d_in[2];   // [2, E]
    const float* W_geo = (const float*)d_in[4];
    const float* b_geo = (const float*)d_in[5];
    const float* emb   = (const float*)d_in[6];
    const float* W_lot = (const float*)d_in[7];
    const float* b_lot = (const float*)d_in[8];
    const float* W_mp[3]  = {(const float*)d_in[9],  (const float*)d_in[11], (const float*)d_in[13]};
    const float* b_mp[3]  = {(const float*)d_in[10], (const float*)d_in[12], (const float*)d_in[14]};
    const float* W_agg = (const float*)d_in[15];
    const float* b_agg = (const float*)d_in[16];
    const float* W_mu  = (const float*)d_in[17];
    const float* b_mu  = (const float*)d_in[18];
    const float* W_var = (const float*)d_in[19];
    const float* b_var = (const float*)d_in[20];

    const int* src = edge_idx;            // edge_index[0]
    const int* dst = edge_idx + N_EDGES;  // edge_index[1]

    size_t off = 0;
    unsigned short* encb   = (unsigned short*)wsalloc(d_ws, off, (size_t)N_NODES * 512 * 2);
    float*          h      = (float*)         wsalloc(d_ws, off, (size_t)N_NODES * 256 * 4);
    unsigned short* hb     = (unsigned short*)wsalloc(d_ws, off, (size_t)N_NODES * 256 * 2);
    float*          s      = (float*)         wsalloc(d_ws, off, (size_t)N_NODES * 256 * 4);
    int*            cnt    = (int*)           wsalloc(d_ws, off, (size_t)N_NODES * 4);
    float*          g      = (float*)         wsalloc(d_ws, off, (size_t)B_GRAPHS * 1024 * 4);
    unsigned short* gb     = (unsigned short*)wsalloc(d_ws, off, (size_t)B_GRAPHS * 1024 * 2);
    float*          latent = (float*)         wsalloc(d_ws, off, (size_t)B_GRAPHS * 256 * 4);
    unsigned short* latb   = (unsigned short*)wsalloc(d_ws, off, (size_t)B_GRAPHS * 256 * 2);
    unsigned short* Wt_lot = (unsigned short*)wsalloc(d_ws, off, (size_t)512 * 256 * 2);
    unsigned short* Wt_mp[3];
    for (int l = 0; l < 3; ++l)
        Wt_mp[l] = (unsigned short*)wsalloc(d_ws, off, (size_t)512 * 256 * 2);
    unsigned short* Wt_agg = (unsigned short*)wsalloc(d_ws, off, (size_t)1024 * 256 * 2);
    unsigned short* Wt_mu  = (unsigned short*)wsalloc(d_ws, off, (size_t)256 * 256 * 2);
    unsigned short* Wt_var = (unsigned short*)wsalloc(d_ws, off, (size_t)256 * 256 * 2);

    const int T = 256;
    auto blocks = [](long long n, int t) { return (int)((n + t - 1) / t); };

    // 1) weight transposes -> bf16 [N][K]
    k_transpose_bf16<<<blocks(512LL * 256, T), T, 0, stream>>>(W_lot, Wt_lot, 512);
    for (int l = 0; l < 3; ++l)
        k_transpose_bf16<<<blocks(512LL * 256, T), T, 0, stream>>>(W_mp[l], Wt_mp[l], 512);
    k_transpose_bf16<<<blocks(1024LL * 256, T), T, 0, stream>>>(W_agg, Wt_agg, 1024);
    k_transpose_bf16<<<blocks(256LL * 256, T), T, 0, stream>>>(W_mu, Wt_mu, 256);
    k_transpose_bf16<<<blocks(256LL * 256, T), T, 0, stream>>>(W_var, Wt_var, 256);

    // 2) enc = [geo-linear | embedding] as bf16
    k_prep_enc<<<blocks((long long)N_NODES * 512, T), T, 0, stream>>>(
        geometry, semantic, W_geo, b_geo, emb, encb);

    // 3) in-degree counts (shared by all 3 msg layers)
    k_zero_i32<<<blocks(N_NODES, T), T, 0, stream>>>(cnt, N_NODES);
    k_count<<<blocks(N_EDGES, T), T, 0, stream>>>(dst, cnt, N_EDGES);

    // 4) h0 = relu(enc @ W_lot[0:512] + b_lot + W_lot[512 + pos])
    k_gemm_dense<16><<<N_NODES / 64, T, 0, stream>>>(
        encb, Wt_lot, b_lot, W_lot + 512 * 256, 1, h, hb);
    k_segmax<<<blocks((long long)B_GRAPHS * 256, T), T, 0, stream>>>(h, g, 0);

    // 5) three message-passing layers
    for (int l = 0; l < 3; ++l) {
        k_zero_f32<<<blocks((long long)N_NODES * 256, T), T, 0, stream>>>(s, N_NODES * 256);
        k_gemm_msg<<<N_EDGES / 64, T, 0, stream>>>(hb, Wt_mp[l], dst, src, s);
        k_msg_fin<<<blocks((long long)N_NODES * 256, T), T, 0, stream>>>(s, cnt, b_mp[l], h, hb);
        k_segmax<<<blocks((long long)B_GRAPHS * 256, T), T, 0, stream>>>(h, g, 256 * (l + 1));
    }

    // 6) readout
    k_f2bf_arr<<<blocks((long long)B_GRAPHS * 1024, T), T, 0, stream>>>(g, gb, B_GRAPHS * 1024);
    k_gemm_dense<32><<<B_GRAPHS / 64, T, 0, stream>>>(
        gb, Wt_agg, b_agg, nullptr, 0, latent, latb);
    float* out_mu  = (float*)d_out;
    float* out_var = (float*)d_out + (size_t)B_GRAPHS * 256;
    k_gemm_dense<8><<<B_GRAPHS / 64, T, 0, stream>>>(
        latb, Wt_mu, b_mu, nullptr, 0, out_mu, nullptr);
    k_gemm_dense<8><<<B_GRAPHS / 64, T, 0, stream>>>(
        latb, Wt_var, b_var, nullptr, 0, out_var, nullptr);
}
